// MSA_850403525285
// MI455X (gfx1250) — compile-verified, matrix operand order checked by reading, not run on hardware
//
#include <hip/hip_runtime.h>
#include <cstdint>
#include <cstddef>

// ---------------------------------------------------------------------------
// MI455X (gfx1250) multi-head self-attention, bf16 WMMA pipeline, v3.
//   - async global->LDS staging (GLOBAL_LOAD_ASYNC_TO_LDS_B128, ASYNCcnt)
//   - double-buffered LDS tiles with deferred s_wait_asynccnt(N) so DMA for
//     tile t+1 overlaps the 16 WMMAs of tile t
//   - weights pre-transposed, V stored [b][h][hd][s] -> all staging is b128
// ---------------------------------------------------------------------------

typedef unsigned short u16;
typedef __bf16    bf16x16 __attribute__((ext_vector_type(16)));
typedef float     floatx8 __attribute__((ext_vector_type(8)));
typedef unsigned int uint4v __attribute__((ext_vector_type(4)));
typedef int       intx4v __attribute__((ext_vector_type(4)));

#define BATCH   4
#define SEQ     2048
#define DIM     1024
#define NHEADS  16
#define HDIM    64
#define N3D     3072
#define MROWS   (BATCH * SEQ)       // 8192
#define SCALE_F 0.125f

#if defined(__has_builtin)
#if __has_builtin(__builtin_amdgcn_global_load_async_to_lds_b128) && \
    __has_builtin(__builtin_amdgcn_s_wait_asynccnt)
#define USE_ASYNC 1
#endif
#endif
#ifndef USE_ASYNC
#define USE_ASYNC 0
#endif

union F8 { floatx8 v; float f[8]; };

static __device__ __forceinline__ u16 f2bf(float f) {
    uint32_t u = __float_as_uint(f);
    uint32_t r = u + 0x7FFFu + ((u >> 16) & 1u);   // round-to-nearest-even
    return (u16)(r >> 16);
}

static __device__ __forceinline__ floatx8 wmma_bf16(bf16x16 a, bf16x16 b, floatx8 c) {
    return __builtin_amdgcn_wmma_f32_16x16x32_bf16(
        false, a, false, b, (short)0, c, false, false);
}

// 16-byte global -> LDS stage (async-to-LDS when available).
static __device__ __forceinline__ void cp16(u16* l, const u16* g) {
#if USE_ASYNC
    __builtin_amdgcn_global_load_async_to_lds_b128(
        (intx4v*)(g), (intx4v*)(l), 0, 0);
#else
    *reinterpret_cast<uint4v*>(l) = *reinterpret_cast<const uint4v*>(g);
#endif
}

// Wait so that all async loads except the `keep` newest have completed
// (ASYNCcnt completes in order), then rendezvous.
static __device__ __forceinline__ void stage_sync(bool more, int /*unused*/) {
#if USE_ASYNC
    if (more) __builtin_amdgcn_s_wait_asynccnt(8);
    else      __builtin_amdgcn_s_wait_asynccnt(0);
#endif
    __syncthreads();
}
static __device__ __forceinline__ void stage_sync4(bool more) {
#if USE_ASYNC
    if (more) __builtin_amdgcn_s_wait_asynccnt(4);
    else      __builtin_amdgcn_s_wait_asynccnt(0);
#endif
    __syncthreads();
}

// 16x32 bf16 fragment from LDS row-major [m][k], leading dim ldk.
// lanes 0-15 : m=lane,    K = {kbase+0..7,  kbase+16..23}
// lanes 16-31: m=lane-16, K = {kbase+8..15, kbase+24..31}
static __device__ __forceinline__ bf16x16 load_frag(const u16* lds, int rowbase,
                                                    int ldk, int kbase, int lane) {
    const u16* p = lds + (size_t)(rowbase + (lane & 15)) * ldk
                       + kbase + ((lane >> 4) << 3);
    union { bf16x16 v; uint4v q[2]; } u;
    u.q[0] = *reinterpret_cast<const uint4v*>(p);
    u.q[1] = *reinterpret_cast<const uint4v*>(p + 16);
    return u.v;
}

// ---------------------------------------------------------------------------
// Kernel 1a: elementwise f32 -> bf16
// ---------------------------------------------------------------------------
__global__ __launch_bounds__(256) void cvt_bf16_kernel(const float* __restrict__ in,
                                                       u16* __restrict__ out, int n) {
    for (int i = blockIdx.x * blockDim.x + threadIdx.x; i < n;
         i += gridDim.x * blockDim.x)
        out[i] = f2bf(in[i]);
}

// ---------------------------------------------------------------------------
// Kernel 1b: tiled transpose f32[K][N] -> bf16[N][K]
// ---------------------------------------------------------------------------
__global__ __launch_bounds__(256) void cvt_t_kernel(const float* __restrict__ in,
                                                    u16* __restrict__ out,
                                                    int K, int N) {
    __shared__ float tile[32][33];
    const int k0 = blockIdx.x * 32, n0 = blockIdx.y * 32;
    const int tx = threadIdx.x & 31, ty = threadIdx.x >> 5;   // ty 0..7
#pragma unroll
    for (int j = 0; j < 4; ++j)
        tile[ty + j * 8][tx] = in[(size_t)(k0 + ty + j * 8) * N + n0 + tx];
    __syncthreads();
#pragma unroll
    for (int j = 0; j < 4; ++j)
        out[(size_t)(n0 + ty + j * 8) * K + k0 + tx] = f2bf(tile[tx][ty + j * 8]);
}

// ---------------------------------------------------------------------------
// GEMM tile engine: C[128x128] block tile, A[M,K] x Bt[N,K] (both bf16,
// Bt pre-transposed).  8 waves, each 32x64 = 2x4 WMMA frags.  K staged 64
// at a time into double-buffered LDS; stage t+1 overlaps compute of t.
// ---------------------------------------------------------------------------
static __device__ void gemm_tile_v3(const u16* __restrict__ A,
                                    const u16* __restrict__ Bt,
                                    int K, floatx8 acc[2][4],
                                    u16* As /*[2][128*64]*/, u16* Bs /*[2][128*64]*/,
                                    int bm, int bn, int tid) {
    const int lane = tid & 31, w = tid >> 5, wm = w >> 1, wn = w & 1;
    const int r = tid >> 3, cg = (tid & 7) << 3;
    const int NT = K / 64;

    // stage tile 0 into buffer 0  (8 async loads per wavefront pass)
#pragma unroll
    for (int p = 0; p < 4; ++p) {
        int row = p * 32 + r;
        cp16(&As[row * 64 + cg], &A[(size_t)(bm * 128 + row) * K + cg]);
        cp16(&Bs[row * 64 + cg], &Bt[(size_t)(bn * 128 + row) * K + cg]);
    }

    for (int kt = 0; kt < NT; ++kt) {
        const int cur = kt & 1, nxt = cur ^ 1;
        const bool more = (kt + 1) < NT;
        if (more) {
#pragma unroll
            for (int p = 0; p < 4; ++p) {
                int row = p * 32 + r;
                cp16(&As[nxt * 8192 + row * 64 + cg],
                     &A[(size_t)(bm * 128 + row) * K + (kt + 1) * 64 + cg]);
                cp16(&Bs[nxt * 8192 + row * 64 + cg],
                     &Bt[(size_t)(bn * 128 + row) * K + (kt + 1) * 64 + cg]);
            }
        }
        stage_sync(more, 8);          // tile kt staged; tile kt+1 in flight

        const u16* Ac = As + cur * 8192;
        const u16* Bc = Bs + cur * 8192;
#pragma unroll
        for (int ks = 0; ks < 2; ++ks) {
            bf16x16 af0 = load_frag(Ac, wm * 32 + 0,  64, ks * 32, lane);
            bf16x16 af1 = load_frag(Ac, wm * 32 + 16, 64, ks * 32, lane);
#pragma unroll
            for (int j = 0; j < 4; ++j) {
                bf16x16 bfr = load_frag(Bc, wn * 64 + j * 16, 64, ks * 32, lane);
                acc[0][j] = wmma_bf16(af0, bfr, acc[0][j]);
                acc[1][j] = wmma_bf16(af1, bfr, acc[1][j]);
            }
        }
        __syncthreads();              // done reading cur before stage kt+2
    }
}

// ---------------------------------------------------------------------------
// Kernel 2: QKV projection.  xb[8192,1024] @ wqkvT -> bf16 scatter into
//   q,k : qkv[(which*4+b)*16+h][s][hd]      (row-major per head)
//   v   : qkv[(8+b)*16+h][hd][s]            (transposed per head)
// ---------------------------------------------------------------------------
__global__ __launch_bounds__(256) void gemm_qkv_kernel(const u16* __restrict__ xb,
                                                       const u16* __restrict__ wbt,
                                                       u16* __restrict__ qkv) {
    __shared__ __align__(16) u16 As[2 * 128 * 64];
    __shared__ __align__(16) u16 Bs[2 * 128 * 64];
    const floatx8 z8 = {0.f,0.f,0.f,0.f,0.f,0.f,0.f,0.f};
    floatx8 acc[2][4] = {{z8, z8, z8, z8}, {z8, z8, z8, z8}};
    gemm_tile_v3(xb, wbt, DIM, acc, As, Bs, blockIdx.x, blockIdx.y, threadIdx.x);

    const int tid = threadIdx.x, lane = tid & 31, w = tid >> 5, wm = w >> 1, wn = w & 1;
    const int hi8 = (lane >> 4) << 3, nn = lane & 15;
#pragma unroll
    for (int i = 0; i < 2; ++i)
#pragma unroll
        for (int j = 0; j < 4; ++j) {
            F8 c; c.v = acc[i][j];
#pragma unroll
            for (int v = 0; v < 8; ++v) {
                int row = blockIdx.x * 128 + wm * 32 + i * 16 + v + hi8;
                int col = blockIdx.y * 128 + wn * 64 + j * 16 + nn;
                int b = row >> 11, s = row & 2047;
                int which = col >> 10, d = col & 1023;
                int h = d >> 6, hd = d & 63;
                size_t idx;
                if (which < 2)   // q, k row-major [s][hd]
                    idx = ((size_t)((which * 4 + b) * 16 + h) * SEQ + s) * HDIM + hd;
                else             // v transposed [hd][s]
                    idx = (((size_t)(8 + b) * 16 + h) * HDIM + hd) * SEQ + s;
                qkv[idx] = f2bf(c.f[v]);
            }
        }
}

// ---------------------------------------------------------------------------
// Kernel 3: flash attention.  One block = 128 query rows of one (b,h).
// Double-buffered K/V tiles; stage kt+1 overlaps softmax + 16 WMMAs of kt.
// ---------------------------------------------------------------------------
__global__ __launch_bounds__(256) void attn_kernel(const u16* __restrict__ qkv,
                                                   u16* __restrict__ attnb) {
    __shared__ __align__(16) u16 Qs[128 * 64];       // 16 KB
    __shared__ __align__(16) u16 Ks[2 * 64 * 64];    // 16 KB [key][hd]
    __shared__ __align__(16) u16 Vt[2 * 64 * 64];    // 16 KB [hd][key]
    __shared__ __align__(16) u16 Ps[128 * 64];       // 16 KB

    const int tid = threadIdx.x, lane = tid & 31, w = tid >> 5;
    const int qb = blockIdx.x & 15;
    const int h  = (blockIdx.x >> 4) & 15;
    const int b  = blockIdx.x >> 8;
    const size_t plane = (size_t)SEQ * HDIM;
    const u16* qp = qkv + ((size_t)(0 + b) * 16 + h) * plane;   // [s][hd]
    const u16* kp = qkv + ((size_t)(4 + b) * 16 + h) * plane;   // [s][hd]
    const u16* vp = qkv + ((size_t)(8 + b) * 16 + h) * plane;   // [hd][s]

    const int r = tid >> 3, cg = (tid & 7) << 3;
#pragma unroll
    for (int p = 0; p < 4; ++p)
        cp16(&Qs[(p * 32 + r) * 64 + cg],
             &qp[(size_t)(qb * 128 + p * 32 + r) * HDIM + cg]);
    // stage K/V tile 0 into buffer 0
#pragma unroll
    for (int p = 0; p < 2; ++p) {
        int rr = p * 32 + r;
        cp16(&Ks[rr * 64 + cg], &kp[(size_t)rr * HDIM + cg]);
        cp16(&Vt[rr * 64 + cg], &vp[(size_t)rr * SEQ + cg]);
    }

    const floatx8 z8 = {0.f,0.f,0.f,0.f,0.f,0.f,0.f,0.f};
    floatx8 O[4] = {z8, z8, z8, z8};
    float m_i[8], l_i[8];
#pragma unroll
    for (int v = 0; v < 8; ++v) { m_i[v] = -3.0e38f; l_i[v] = 0.f; }

    const int hi8 = (lane >> 4) << 3, nn = lane & 15;
    const int NT = SEQ / 64;

    for (int kt = 0; kt < NT; ++kt) {
        const int cur = kt & 1, nxt = cur ^ 1;
        const bool more = (kt + 1) < NT;
        if (more) {
#pragma unroll
            for (int p = 0; p < 2; ++p) {
                int rr = p * 32 + r;
                cp16(&Ks[nxt * 4096 + rr * 64 + cg],
                     &kp[(size_t)((kt + 1) * 64 + rr) * HDIM + cg]);
                cp16(&Vt[nxt * 4096 + rr * 64 + cg],
                     &vp[(size_t)rr * SEQ + (kt + 1) * 64 + cg]);
            }
        }
        stage_sync4(more);            // tile kt (and Q) staged; kt+1 in flight

        const u16* Kc = Ks + cur * 4096;
        const u16* Vc = Vt + cur * 4096;

        // S = Q @ K^T
        floatx8 sacc[4] = {z8, z8, z8, z8};
#pragma unroll
        for (int ks = 0; ks < 2; ++ks) {
            bf16x16 af = load_frag(Qs, w * 16, 64, ks * 32, lane);
#pragma unroll
            for (int nt = 0; nt < 4; ++nt) {
                bf16x16 bfr = load_frag(Kc, nt * 16, 64, ks * 32, lane);
                sacc[nt] = wmma_bf16(af, bfr, sacc[nt]);
            }
        }

        // online softmax over the 64-key tile
        F8 st[4];
#pragma unroll
        for (int nt = 0; nt < 4; ++nt) st[nt].v = sacc[nt];
        float al[8];
#pragma unroll
        for (int v = 0; v < 8; ++v) {
#pragma unroll
            for (int nt = 0; nt < 4; ++nt) st[nt].f[v] *= SCALE_F;
            float mx = fmaxf(fmaxf(st[0].f[v], st[1].f[v]),
                             fmaxf(st[2].f[v], st[3].f[v]));
            mx = fmaxf(mx, __shfl_xor(mx, 1, 32));
            mx = fmaxf(mx, __shfl_xor(mx, 2, 32));
            mx = fmaxf(mx, __shfl_xor(mx, 4, 32));
            mx = fmaxf(mx, __shfl_xor(mx, 8, 32));
            float mn = fmaxf(m_i[v], mx);
            float alpha = __expf(m_i[v] - mn);
            float rs = 0.f;
#pragma unroll
            for (int nt = 0; nt < 4; ++nt) {
                float pv = __expf(st[nt].f[v] - mn);
                st[nt].f[v] = pv;
                rs += pv;
            }
            rs += __shfl_xor(rs, 1, 32);
            rs += __shfl_xor(rs, 2, 32);
            rs += __shfl_xor(rs, 4, 32);
            rs += __shfl_xor(rs, 8, 32);
            l_i[v] = l_i[v] * alpha + rs;
            m_i[v] = mn;
            al[v]  = alpha;
        }
#pragma unroll
        for (int nt = 0; nt < 4; ++nt) {
            F8 o; o.v = O[nt];
#pragma unroll
            for (int v = 0; v < 8; ++v) o.f[v] *= al[v];
            O[nt] = o.v;
        }
        // spill P (bf16) to this wave's private rows of Ps; same-wave DS
        // ordering guarantees the following frag loads see the stores.
#pragma unroll
        for (int nt = 0; nt < 4; ++nt)
#pragma unroll
            for (int v = 0; v < 8; ++v)
                Ps[(w * 16 + v + hi8) * 64 + nt * 16 + nn] = f2bf(st[nt].f[v]);

        // O += P @ V
#pragma unroll
        for (int ks = 0; ks < 2; ++ks) {
            bf16x16 af = load_frag(Ps, w * 16, 64, ks * 32, lane);
#pragma unroll
            for (int nt = 0; nt < 4; ++nt) {
                bf16x16 bfr = load_frag(Vc, nt * 16, 64, ks * 32, lane);
                O[nt] = wmma_bf16(af, bfr, O[nt]);
            }
        }
        __syncthreads();              // done reading cur before stage kt+2
    }

    // normalize and write attn output as bf16 matrix [B*S, D]
#pragma unroll
    for (int nt = 0; nt < 4; ++nt) {
        F8 o; o.v = O[nt];
#pragma unroll
        for (int v = 0; v < 8; ++v) {
            int srow = qb * 128 + w * 16 + v + hi8;
            float val = o.f[v] / l_i[v];
            attnb[(size_t)(b * SEQ + srow) * DIM + h * HDIM + nt * 16 + nn] =
                f2bf(val);
        }
    }
}

// ---------------------------------------------------------------------------
// Kernel 4: output projection.  attn[8192,1024] @ wprojT + bias -> f32 out
// ---------------------------------------------------------------------------
__global__ __launch_bounds__(256) void gemm_proj_kernel(const u16* __restrict__ ab,
                                                        const u16* __restrict__ wbt,
                                                        const float* __restrict__ bias,
                                                        float* __restrict__ out) {
    __shared__ __align__(16) u16 As[2 * 128 * 64];
    __shared__ __align__(16) u16 Bs[2 * 128 * 64];
    const floatx8 z8 = {0.f,0.f,0.f,0.f,0.f,0.f,0.f,0.f};
    floatx8 acc[2][4] = {{z8, z8, z8, z8}, {z8, z8, z8, z8}};
    gemm_tile_v3(ab, wbt, DIM, acc, As, Bs, blockIdx.x, blockIdx.y, threadIdx.x);

    const int tid = threadIdx.x, lane = tid & 31, w = tid >> 5, wm = w >> 1, wn = w & 1;
    const int hi8 = (lane >> 4) << 3, nn = lane & 15;
#pragma unroll
    for (int i = 0; i < 2; ++i)
#pragma unroll
        for (int j = 0; j < 4; ++j) {
            F8 c; c.v = acc[i][j];
#pragma unroll
            for (int v = 0; v < 8; ++v) {
                int row = blockIdx.x * 128 + wm * 32 + i * 16 + v + hi8;
                int col = blockIdx.y * 128 + wn * 64 + j * 16 + nn;
                out[(size_t)row * DIM + col] = c.f[v] + bias[col];
            }
        }
}

// ---------------------------------------------------------------------------
// Launch
// ---------------------------------------------------------------------------
extern "C" void kernel_launch(void* const* d_in, const int* in_sizes, int n_in,
                              void* d_out, int out_size, void* d_ws, size_t ws_size,
                              hipStream_t stream) {
    const float* x      = (const float*)d_in[0];
    const float* w_qkv  = (const float*)d_in[1];
    const float* w_proj = (const float*)d_in[2];
    const float* b_proj = (const float*)d_in[3];
    float* out = (float*)d_out;

    u16* ws      = (u16*)d_ws;
    u16* xb      = ws;                                    // 8192*1024
    u16* wqkvT   = xb    + (size_t)MROWS * DIM;           // [3072][1024]
    u16* wprojT  = wqkvT + (size_t)DIM * N3D;             // [1024][1024]
    u16* qkvb    = wprojT + (size_t)DIM * DIM;            // 3*4*16*2048*64
    u16* attnb   = qkvb  + (size_t)3 * BATCH * NHEADS * SEQ * HDIM; // 8192*1024

    cvt_bf16_kernel<<<4096, 256, 0, stream>>>(x, xb, MROWS * DIM);
    cvt_t_kernel<<<dim3(DIM / 32, N3D / 32), 256, 0, stream>>>(w_qkv, wqkvT, DIM, N3D);
    cvt_t_kernel<<<dim3(DIM / 32, DIM / 32), 256, 0, stream>>>(w_proj, wprojT, DIM, DIM);

    gemm_qkv_kernel<<<dim3(MROWS / 128, N3D / 128), 256, 0, stream>>>(xb, wqkvT, qkvb);
    attn_kernel<<<BATCH * NHEADS * (SEQ / 128), 256, 0, stream>>>(qkvb, attnb);
    gemm_proj_kernel<<<dim3(MROWS / 128, DIM / 128), 256, 0, stream>>>(attnb, wprojT,
                                                                      b_proj, out);
}
